// GRU_4_FC_3_40175124086989
// MI455X (gfx1250) — matrix operand order repaired and checked by reading, compiled, not run on hardware
//
#include <hip/hip_runtime.h>
#include <hip/hip_bf16.h>

typedef __attribute__((ext_vector_type(16))) __bf16 v16bf;
typedef __attribute__((ext_vector_type(8)))  float  v8f;

union BF16x16 {
    v16bf v;
    unsigned short u[16];
};

__device__ __forceinline__ unsigned short f2bfbits(float f) {
    unsigned int u = __builtin_bit_cast(unsigned int, f);
    unsigned int r = u + 0x7FFFu + ((u >> 16) & 1u);
    return (unsigned short)(r >> 16);
}

// Branch-free fast transcendentals on the v_exp_f32 / v_rcp_f32 path.
__device__ __forceinline__ float sigmoidf_(float x) {
    const float e = __builtin_amdgcn_exp2f(-1.442695041f * x);
    return __builtin_amdgcn_rcpf(1.0f + e);
}
__device__ __forceinline__ float tanhf_(float x) {
    const float e = __builtin_amdgcn_exp2f(2.885390082f * x);  // exp(2x)
    return 1.0f - 2.0f * __builtin_amdgcn_rcpf(e + 1.0f);      // (e-1)/(e+1)
}

// Load one 32x16 bf16 B-fragment pair (K=0..31 and K=32..63) for output tile
// `tile` from an LDS row-major [192][64] weight image.
// B-fragment layout (16-bit, 32x16): lane = {half,n}; element e -> K = 32*g + 16*(e>>3) + 8*half + (e&7).
__device__ __forceinline__ void load_bfrag(const unsigned short* Wl, int tile, int half, int ln,
                                           BF16x16& b0, BF16x16& b1) {
    const unsigned short* p = Wl + (tile * 16 + ln) * 64 + 8 * half;
#pragma unroll
    for (int e = 0; e < 16; ++e) { const int k = (e & 7) + 16 * (e >> 3); b0.u[e] = p[k]; b1.u[e] = p[32 + k]; }
}

// C(16x16,f32) = bias + A(16x64,bf16) * B(64x16,bf16), fragments in registers.
__device__ __forceinline__ v8f tile_mm(float bias, v16bf A0, v16bf A1, v16bf B0, v16bf B1) {
    v8f c;
#pragma unroll
    for (int v = 0; v < 8; ++v) c[v] = bias;
    c = __builtin_amdgcn_wmma_f32_16x16x32_bf16(false, A0, false, B0, (short)0, c, false, false);
    c = __builtin_amdgcn_wmma_f32_16x16x32_bf16(false, A1, false, B1, (short)0, c, false, false);
    return c;
}

// GRU layer scan.  Grid: B/16 workgroups, 128 threads (4 waves).  Each WG owns a
// 16-row batch tile across all T steps; wave w computes gate-column group w
// (cols 16w..16w+15 of H=64; gate tiles w, w+4, w+8 of the 192-wide projection).
// IN_W==0: input is previous layer's bf16 output [B,T,64], x-projection via WMMA.
// IN_W>0 : raw f32 input [B,T,IN_W], x-projection via VALU.
template <int IN_W>
__global__ __launch_bounds__(128) void gru_scan(
    const float* __restrict__ xraw, const unsigned short* __restrict__ yin,
    const float* __restrict__ wih0, const float* __restrict__ wih,
    const float* __restrict__ whh, const float* __restrict__ bih,
    const float* __restrict__ bhh, unsigned short* __restrict__ yout,
    float* __restrict__ hlast) {
    constexpr int T = 256, H = 64;
    constexpr int WSZ = 192 * 64;

    __shared__ unsigned short lds_w[(IN_W == 0 ? 2 : 1) * WSZ];
    __shared__ unsigned short lds_h[16 * 64];

    const int tid  = threadIdx.x;
    const int wave = tid >> 5;        // 0..3  -> gate column group jg
    const int lane = tid & 31;
    const int half = (lane >> 4) & 1;
    const int ln   = lane & 15;
    const int jg   = wave;
    const int m0   = blockIdx.x * 16;

    // Stage weights -> LDS as bf16 (Whh always; Wih for WMMA layers).
    for (int i = tid; i < WSZ; i += 128) {
        lds_w[i] = f2bfbits(whh[i]);
        if constexpr (IN_W == 0) lds_w[WSZ + i] = f2bfbits(wih[i]);
    }
    __syncthreads();
    const unsigned short* lds_whh = lds_w;
    const unsigned short* lds_wih = lds_w + (IN_W == 0 ? WSZ : 0);

    // Hoist the loop-invariant weight B-fragments into registers:
    // 6 tiles (r/z/n for Whh, and for Wih on WMMA layers) x 2 K-groups.
    BF16x16 Bhr0, Bhr1, Bhz0, Bhz1, Bhn0, Bhn1;
    load_bfrag(lds_whh, jg,     half, ln, Bhr0, Bhr1);
    load_bfrag(lds_whh, jg + 4, half, ln, Bhz0, Bhz1);
    load_bfrag(lds_whh, jg + 8, half, ln, Bhn0, Bhn1);
    BF16x16 Bxr0, Bxr1, Bxz0, Bxz1, Bxn0, Bxn1;
    if constexpr (IN_W == 0) {
        load_bfrag(lds_wih, jg,     half, ln, Bxr0, Bxr1);
        load_bfrag(lds_wih, jg + 4, half, ln, Bxz0, Bxz1);
        load_bfrag(lds_wih, jg + 8, half, ln, Bxn0, Bxn1);
    }

    // Per-lane gate biases (C layout: column = tile*16 + ln for both lane halves).
    const float bihr = bih[(jg)      * 16 + ln], bhhr = bhh[(jg)      * 16 + ln];
    const float bihz = bih[(jg + 4)  * 16 + ln], bhhz = bhh[(jg + 4)  * 16 + ln];
    const float bihn = bih[(jg + 8)  * 16 + ln], bhhn = bhh[(jg + 8)  * 16 + ln];

    // Layer-0 input weights per lane.
    float w0r[4], w0z[4], w0n[4];
    if constexpr (IN_W > 0) {
#pragma unroll
        for (int i = 0; i < IN_W; ++i) {
            w0r[i] = wih0[((jg)     * 16 + ln) * IN_W + i];
            w0z[i] = wih0[((jg + 4) * 16 + ln) * IN_W + i];
            w0n[i] = wih0[((jg + 8) * 16 + ln) * IN_W + i];
        }
    }

    // h state: C-layout f32 (this wave's 16 cols) + full A-layout bf16 fragments.
    float hold[8];
#pragma unroll
    for (int v = 0; v < 8; ++v) hold[v] = 0.0f;
    BF16x16 hA0, hA1;
#pragma unroll
    for (int e = 0; e < 16; ++e) { hA0.u[e] = 0; hA1.u[e] = 0; }

    const int col = jg * 16 + ln;

    for (int t = 0; t < T; ++t) {
        // --- gather x_t ---
        BF16x16 xA0, xA1;
        float xv[8][IN_W > 0 ? IN_W : 1];
        if constexpr (IN_W == 0) {
            // A-layout (16-bit 16x32): m = ln; element e -> K = 16*grp + 8*half + 2*jj + pos.
            const unsigned short* xrow = yin + ((size_t)(m0 + ln) * T + t) * (size_t)H;
#pragma unroll
            for (int e = 0; e < 16; ++e) {
                const int jj = (e >> 1) & 3, grp = (e >> 3) & 1, pos = e & 1;
                const int k = 16 * grp + 8 * half + 2 * jj + pos;
                xA0.u[e] = xrow[k];
                xA1.u[e] = xrow[32 + k];
            }
            if (t + 1 < T) __builtin_prefetch(xrow + H, 0, 0);  // next step is contiguous
        } else {
#pragma unroll
            for (int v = 0; v < 8; ++v) {
                const int m = v + 8 * half;
                const float* p = xraw + ((size_t)(m0 + m) * T + t) * (size_t)IN_W;
#pragma unroll
                for (int i = 0; i < IN_W; ++i) xv[v][i] = p[i];
            }
        }

        // --- projections (WMMA, all operands in registers) ---
        v8f hr = tile_mm(bhhr, hA0.v, hA1.v, Bhr0.v, Bhr1.v);
        v8f hz = tile_mm(bhhz, hA0.v, hA1.v, Bhz0.v, Bhz1.v);
        v8f hn = tile_mm(bhhn, hA0.v, hA1.v, Bhn0.v, Bhn1.v);
        v8f xr, xz, xn;
        if constexpr (IN_W == 0) {
            xr = tile_mm(bihr, xA0.v, xA1.v, Bxr0.v, Bxr1.v);
            xz = tile_mm(bihz, xA0.v, xA1.v, Bxz0.v, Bxz1.v);
            xn = tile_mm(bihn, xA0.v, xA1.v, Bxn0.v, Bxn1.v);
        } else {
#pragma unroll
            for (int v = 0; v < 8; ++v) {
                float a = bihr, b = bihz, c = bihn;
#pragma unroll
                for (int i = 0; i < IN_W; ++i) {
                    a += xv[v][i] * w0r[i];
                    b += xv[v][i] * w0z[i];
                    c += xv[v][i] * w0n[i];
                }
                xr[v] = a; xz[v] = b; xn[v] = c;
            }
        }

        // --- gates + state update (all accs share the C layout) ---
#pragma unroll
        for (int v = 0; v < 8; ++v) {
            const float r = sigmoidf_(xr[v] + hr[v]);
            const float z = sigmoidf_(xz[v] + hz[v]);
            const float n = tanhf_(xn[v] + r * hn[v]);
            hold[v] = (1.0f - z) * n + z * hold[v];
        }

        // --- publish h_new: LDS stage (for A-frag rebuild) + global bf16 output ---
#pragma unroll
        for (int v = 0; v < 8; ++v) {
            const int m = v + 8 * half;
            const unsigned short hb = f2bfbits(hold[v]);
            lds_h[m * 64 + col] = hb;
            yout[((size_t)(m0 + m) * T + t) * (size_t)H + col] = hb;
            if (t == T - 1) hlast[(size_t)(m0 + m) * H + col] = hold[v];
        }
        __syncthreads();
        // rebuild A-layout h fragments (cross-lane transpose via LDS)
#pragma unroll
        for (int e = 0; e < 16; ++e) {
            const int jj = (e >> 1) & 3, grp = (e >> 3) & 1, pos = e & 1;
            const int k = 16 * grp + 8 * half + 2 * jj + pos;
            hA0.u[e] = lds_h[ln * 64 + k];
            hA1.u[e] = lds_h[ln * 64 + 32 + k];
        }
        __syncthreads();
    }
}

// ---------------- MLP head (tiny; VALU) ----------------

__global__ void fc_head1(const float* __restrict__ sh, const float* __restrict__ ph,
                         const float* __restrict__ W, const float* __restrict__ b,
                         const float* __restrict__ g, const float* __restrict__ be,
                         const float* __restrict__ mn, const float* __restrict__ vr,
                         float* __restrict__ out) {
    const int idx = blockIdx.x * 256 + threadIdx.x;
    if (idx >= 2048 * 128) return;
    const int bb = idx >> 7, o = idx & 127;
    const float* wr = W + o * 128;
    const float* s = sh + bb * 64;
    const float* p = ph + bb * 64;
    float acc = b[o];
#pragma unroll 4
    for (int k = 0; k < 64; ++k) acc += s[k] * wr[k];
#pragma unroll 4
    for (int k = 0; k < 64; ++k) acc += p[k] * wr[64 + k];
    const float x = g[o] * (acc - mn[o]) * rsqrtf(vr[o] + 1e-5f) + be[o];
    out[idx] = fmaxf(x, 0.0f);
}

__global__ void fc_head2(const float* __restrict__ xin,
                         const float* __restrict__ W, const float* __restrict__ b,
                         const float* __restrict__ g, const float* __restrict__ be,
                         const float* __restrict__ mn, const float* __restrict__ vr,
                         float* __restrict__ out) {
    const int idx = blockIdx.x * 256 + threadIdx.x;
    if (idx >= 2048 * 128) return;
    const int bb = idx >> 7, o = idx & 127;
    const float* wr = W + o * 128;
    const float* x = xin + bb * 128;
    float acc = b[o];
#pragma unroll 4
    for (int k = 0; k < 128; ++k) acc += x[k] * wr[k];
    const float y = g[o] * (acc - mn[o]) * rsqrtf(vr[o] + 1e-5f) + be[o];
    out[idx] = fmaxf(y, 0.0f);
}

__global__ void fc_head3(const float* __restrict__ xin,
                         const float* __restrict__ W, const float* __restrict__ b,
                         float* __restrict__ out) {
    const int idx = blockIdx.x * 256 + threadIdx.x;
    if (idx >= 2048 * 2) return;
    const int bb = idx >> 1, o = idx & 1;
    const float* wr = W + o * 128;
    const float* x = xin + bb * 128;
    float acc = b[o];
#pragma unroll 4
    for (int k = 0; k < 128; ++k) acc += x[k] * wr[k];
    out[idx] = acc;
}

extern "C" void kernel_launch(void* const* d_in, const int* in_sizes, int n_in,
                              void* d_out, int out_size, void* d_ws, size_t ws_size,
                              hipStream_t stream) {
    (void)in_sizes; (void)n_in; (void)out_size; (void)ws_size;
    const float* sent  = (const float*)d_in[0];
    const float* price = (const float*)d_in[1];
    const float* sW0   = (const float*)d_in[2];
    const float* sWih  = (const float*)d_in[3];
    const float* sWhh  = (const float*)d_in[4];
    const float* sbih  = (const float*)d_in[5];
    const float* sbhh  = (const float*)d_in[6];
    const float* pW0   = (const float*)d_in[7];
    const float* pWih  = (const float*)d_in[8];
    const float* pWhh  = (const float*)d_in[9];
    const float* pbih  = (const float*)d_in[10];
    const float* pbhh  = (const float*)d_in[11];
    const float* fc1W  = (const float*)d_in[12];
    const float* fc1b  = (const float*)d_in[13];
    const float* g1    = (const float*)d_in[14];
    const float* be1   = (const float*)d_in[15];
    const float* mn1   = (const float*)d_in[16];
    const float* vr1   = (const float*)d_in[17];
    const float* fc2W  = (const float*)d_in[18];
    const float* fc2b  = (const float*)d_in[19];
    const float* g2    = (const float*)d_in[20];
    const float* be2   = (const float*)d_in[21];
    const float* mn2   = (const float*)d_in[22];
    const float* vr2   = (const float*)d_in[23];
    const float* fc3W  = (const float*)d_in[24];
    const float* fc3b  = (const float*)d_in[25];

    constexpr int Bn = 2048, Tn = 256, Hn = 64;
    constexpr size_t YE = (size_t)Bn * Tn * Hn;  // bf16 elements per activation buffer
    unsigned short* yA = (unsigned short*)d_ws;
    unsigned short* yB = yA + YE;
    float* shl = (float*)(yB + YE);
    float* phl = shl + (size_t)Bn * Hn;
    float* t1  = phl + (size_t)Bn * Hn;
    float* t2  = t1 + (size_t)Bn * 128;

    const int J = 192 * 64;
    dim3 grd(Bn / 16), blk(128);

    // sentiment stack (input width 1)
    gru_scan<1><<<grd, blk, 0, stream>>>(sent, nullptr, sW0, nullptr,
                                         sWhh, sbih, sbhh, yA, shl);
    gru_scan<0><<<grd, blk, 0, stream>>>(nullptr, yA, nullptr, sWih,
                                         sWhh + J, sbih + 192, sbhh + 192, yB, shl);
    gru_scan<0><<<grd, blk, 0, stream>>>(nullptr, yB, nullptr, sWih + J,
                                         sWhh + 2 * J, sbih + 384, sbhh + 384, yA, shl);
    gru_scan<0><<<grd, blk, 0, stream>>>(nullptr, yA, nullptr, sWih + 2 * J,
                                         sWhh + 3 * J, sbih + 576, sbhh + 576, yB, shl);
    // price stack (input width 4), reuse ping-pong buffers
    gru_scan<4><<<grd, blk, 0, stream>>>(price, nullptr, pW0, nullptr,
                                         pWhh, pbih, pbhh, yA, phl);
    gru_scan<0><<<grd, blk, 0, stream>>>(nullptr, yA, nullptr, pWih,
                                         pWhh + J, pbih + 192, pbhh + 192, yB, phl);
    gru_scan<0><<<grd, blk, 0, stream>>>(nullptr, yB, nullptr, pWih + J,
                                         pWhh + 2 * J, pbih + 384, pbhh + 384, yA, phl);
    gru_scan<0><<<grd, blk, 0, stream>>>(nullptr, yA, nullptr, pWih + 2 * J,
                                         pWhh + 3 * J, pbih + 576, pbhh + 576, yB, phl);

    // MLP head
    fc_head1<<<(2048 * 128) / 256, 256, 0, stream>>>(shl, phl, fc1W, fc1b, g1, be1, mn1, vr1, t1);
    fc_head2<<<(2048 * 128) / 256, 256, 0, stream>>>(t1, fc2W, fc2b, g2, be2, mn2, vr2, t2);
    fc_head3<<<(2048 * 2 + 255) / 256, 256, 0, stream>>>(t2, fc3W, fc3b, (float*)d_out);
}
